// SJN_Meta_2_55276229099613
// MI455X (gfx1250) — matrix operand order repaired and checked by reading, not compiled, not run on hardware
//
#include <hip/hip_runtime.h>
#include <math.h>

// MI455X (gfx1250) implementation of the 3-round MetaLayer GNN.
//
// bf16 activations + V_WMMA_F32_16X16X32_BF16; BatchNorm folded into per-layer
// bf16 weights (stats via one streaming pass with LDS float atomics);
// LeakyReLU fused in the GEMM epilogue; segment_sum fused as global f32
// atomics in the last node1 GEMM epilogue.  This revision streamlines the
// GEMM epilogue (wave-uniform tail branch, immediate-offset stores, select
// instead of per-element EXEC branches) and adds M-tiling so each B-fragment
// load feeds MT WMMAs.

typedef __bf16 bf16;
typedef __attribute__((ext_vector_type(8)))  bf16  bf16x8;
typedef __attribute__((ext_vector_type(16))) bf16  bf16x16;
typedef __attribute__((ext_vector_type(8)))  float v8f;

#define LEAKK  0.1f
#define BN_EPS 1e-5f
#define SSTR   320   // stats stride (sumsq stored at +SSTR)

// ---------------------------------------------------------------------------
// Elementwise / gather kernels
// ---------------------------------------------------------------------------
__global__ void k_cvt_pad(const float* __restrict__ src, bf16* __restrict__ dst,
                          long long M, int K, int W) {
  long long i = (long long)blockIdx.x * blockDim.x + threadIdx.x;
  if (i >= M * (long long)W) return;
  int c = (int)(i % W);
  long long r = i / W;
  dst[i] = (c < K) ? (bf16)src[r * K + c] : (bf16)0.0f;
}

// concat([x[row], x[col], edge_attr]) -> [E, 64] bf16 (51 valid, pad 0)
__global__ void k_concat_edge(const bf16* __restrict__ xbf, const bf16* __restrict__ eabf,
                              const int* __restrict__ row, const int* __restrict__ col,
                              bf16* __restrict__ out, int E) {
  int e = blockIdx.x * blockDim.x + threadIdx.x;
  if (e >= E) return;
  const bf16* xs = xbf + (size_t)row[e] * 16;
  const bf16* xd = xbf + (size_t)col[e] * 16;
  const bf16* ea = eabf + (size_t)e * 32;
  bf16* o = out + (size_t)e * 64;
#pragma unroll
  for (int i = 0; i < 16; ++i) o[i] = xs[i];
#pragma unroll
  for (int i = 0; i < 16; ++i) o[16 + i] = xd[i];
#pragma unroll
  for (int i = 0; i < 19; ++i) o[32 + i] = ea[i];
#pragma unroll
  for (int i = 51; i < 64; ++i) o[i] = (bf16)0.0f;
}

// concat([x[row], edge_attr]) -> [E, 64] bf16 (35 valid, pad 0)
__global__ void k_concat_node1(const bf16* __restrict__ xbf, const bf16* __restrict__ eabf,
                               const int* __restrict__ row,
                               bf16* __restrict__ out, int E) {
  int e = blockIdx.x * blockDim.x + threadIdx.x;
  if (e >= E) return;
  const bf16* xs = xbf + (size_t)row[e] * 16;
  const bf16* ea = eabf + (size_t)e * 32;
  bf16* o = out + (size_t)e * 64;
#pragma unroll
  for (int i = 0; i < 16; ++i) o[i] = xs[i];
#pragma unroll
  for (int i = 0; i < 19; ++i) o[16 + i] = ea[i];
#pragma unroll
  for (int i = 35; i < 64; ++i) o[i] = (bf16)0.0f;
}

// concat([x, agg]) -> [N, SPAD] bf16 (16+L valid, pad 0)
__global__ void k_concat_node2(const bf16* __restrict__ xbf, const float* __restrict__ agg,
                               bf16* __restrict__ out, int N, int L, int SPAD) {
  int n = blockIdx.x * blockDim.x + threadIdx.x;
  if (n >= N) return;
  bf16* o = out + (size_t)n * SPAD;
  const bf16* xs = xbf + (size_t)n * 16;
#pragma unroll
  for (int i = 0; i < 16; ++i) o[i] = xs[i];
  const float* a = agg + (size_t)n * L;
  for (int i = 0; i < L; ++i) o[16 + i] = (bf16)a[i];
  for (int i = 16 + L; i < SPAD; ++i) o[i] = (bf16)0.0f;
}

// ---------------------------------------------------------------------------
// Column statistics for BatchNorm folding: st[c]=sum, st[SSTR+c]=sum of squares
// ---------------------------------------------------------------------------
__global__ void k_colstats(const bf16* __restrict__ A, long long M, int W,
                           float* __restrict__ st) {
  __shared__ float ls[2 * SSTR];
  for (int i = threadIdx.x; i < 2 * SSTR; i += blockDim.x) ls[i] = 0.0f;
  __syncthreads();
  long long total = M * (long long)W;
  long long stride = (long long)gridDim.x * blockDim.x;
  for (long long i = (long long)blockIdx.x * blockDim.x + threadIdx.x; i < total; i += stride) {
    int c = (int)(i % W);
    float v = (float)A[i];
    atomicAdd(&ls[c], v);            // ds_add_f32
    atomicAdd(&ls[SSTR + c], v * v); // ds_add_f32
  }
  __syncthreads();
  for (int i = threadIdx.x; i < W; i += blockDim.x) {
    atomicAdd(&st[i], ls[i]);
    atomicAdd(&st[SSTR + i], ls[SSTR + i]);
  }
}

// Fold BN into weights:  W'[o,k] = W[o,k]*g[k]*rsqrt(var+eps)   (bf16, K padded)
//                        c'[o]   = c[o] + sum_k W[o,k]*(b[k] - mu[k]*scale[k])
__global__ void k_prep(const float* __restrict__ st, float invM,
                       const float* __restrict__ g, const float* __restrict__ b,
                       const float* __restrict__ W, const float* __restrict__ c,
                       int K, int KPAD, int NOUT, int NTOT,
                       bf16* __restrict__ Wb, float* __restrict__ cb) {
  int o = blockIdx.x * blockDim.x + threadIdx.x;
  if (o >= NTOT) return;
  bf16* wd = Wb + (size_t)o * KPAD;
  if (o >= NOUT) {
    for (int k = 0; k < KPAD; ++k) wd[k] = (bf16)0.0f;
    cb[o] = 0.0f;
    return;
  }
  const float* wr = W + (size_t)o * K;
  float acc = 0.0f;
  for (int k = 0; k < KPAD; ++k) {
    bf16 wv = (bf16)0.0f;
    if (k < K) {
      float mu  = st[k] * invM;
      float var = st[SSTR + k] * invM - mu * mu;
      float s   = g[k] * rsqrtf(var + BN_EPS);
      float sh  = b[k] - mu * s;
      float wf  = wr[k];
      wv = (bf16)(wf * s);
      acc += wf * sh;
    }
    wd[k] = wv;
  }
  cb[o] = c[o] + acc;
}

// ---------------------------------------------------------------------------
// WMMA GEMM:  out[M, NT*16] = act( A[M, KT*32](bf16) @ Wb[NT*16, KT*32]^T + cb )
// Each wave owns MT 16-row strips; A-fragments cached in registers across all
// N tiles; each B-fragment load feeds MT WMMAs.  SCATTER variant fuses
// segment_sum: atomicAdd into agg[sidx[r]*NOUT + n].
// ---------------------------------------------------------------------------
template <int MT, int KT, int NT, int NOUT, bool RELU, bool SCATTER>
__global__ __launch_bounds__(256) void k_gemm(
    const bf16* __restrict__ A, int M,
    const bf16* __restrict__ Wb, const float* __restrict__ cb,
    bf16* __restrict__ out, const int* __restrict__ sidx, float* __restrict__ agg) {
  constexpr int KW = KT * 32;
  constexpr int NW = NT * 16;
  constexpr bool FULLCOLS = (NOUT == NW);
  const int lane = threadIdx.x & 31;
  const int wave = threadIdx.x >> 5;
  const int tile = blockIdx.x * 8 + wave;      // MT*16-row super-strip
  const int m0 = tile * (MT * 16);
  if (m0 >= M) return;

  // A fragments: lanes 0-15 row M=lane hold K {0..7,16..23}; lanes 16-31 {8..15,24..31}
  const int akoff = (lane & 16) ? 8 : 0;
  bf16x16 af[MT][KT];
#pragma unroll
  for (int mt = 0; mt < MT; ++mt) {
    int arow = m0 + mt * 16 + (lane & 15);
    if (arow > M - 1) arow = M - 1;            // tail clamp (stores are guarded)
    const bf16* ap = A + (size_t)arow * KW + akoff;
#pragma unroll
    for (int kt = 0; kt < KT; ++kt) {
      bf16x8 lo = *(const bf16x8*)(ap + kt * 32);
      bf16x8 hi = *(const bf16x8*)(ap + kt * 32 + 16);
      af[mt][kt] = __builtin_shufflevector(lo, hi, 0, 1, 2, 3, 4, 5, 6, 7,
                                           8, 9, 10, 11, 12, 13, 14, 15);
    }
  }

  const int bkoff = (lane & 16) ? 8 : 0;       // B: same K order as A ({0..7,16..23} / {8..15,24..31})
  const int rsub = (lane & 16) ? 8 : 0;        // row sub-offset of this half-wave

  for (int nt = 0; nt < NT; ++nt) {
    const int ncol = nt * 16 + (lane & 15);
    const bf16* bp = Wb + (size_t)ncol * KW + bkoff;
    v8f acc[MT];
#pragma unroll
    for (int mt = 0; mt < MT; ++mt) acc[mt] = (v8f){};
#pragma unroll
    for (int kt = 0; kt < KT; ++kt) {
      bf16x8 blo = *(const bf16x8*)(bp + kt * 32);
      bf16x8 bhi = *(const bf16x8*)(bp + kt * 32 + 16);
      bf16x16 bfrag = __builtin_shufflevector(blo, bhi, 0, 1, 2, 3, 4, 5, 6, 7,
                                              8, 9, 10, 11, 12, 13, 14, 15);
#pragma unroll
      for (int mt = 0; mt < MT; ++mt)
        acc[mt] = __builtin_amdgcn_wmma_f32_16x16x32_bf16(
            false, af[mt][kt], false, bfrag, (short)0, acc[mt], false, false);
    }
    const float bias = cb[ncol];
    const bool colvalid = FULLCOLS ? true : (ncol < NOUT);

#pragma unroll
    for (int mt = 0; mt < MT; ++mt) {
      const int rbase = m0 + mt * 16 + rsub;
      float vv[8];
#pragma unroll
      for (int i = 0; i < 8; ++i) {
        float v = acc[mt][i] + bias;
        if (RELU) v = (v > 0.0f) ? v : (LEAKK * v);
        vv[i] = (FULLCOLS || colvalid) ? v : 0.0f;  // zero-fill pad columns
      }
      const bool fullstrip = (m0 + mt * 16 + 16 <= M);  // wave-uniform
      if (SCATTER) {
        if (colvalid) {
          if (fullstrip) {
#pragma unroll
            for (int i = 0; i < 8; ++i)
              atomicAdd(agg + (size_t)sidx[rbase + i] * NOUT + ncol, vv[i]);
          } else {
#pragma unroll
            for (int i = 0; i < 8; ++i)
              if (rbase + i < M)
                atomicAdd(agg + (size_t)sidx[rbase + i] * NOUT + ncol, vv[i]);
          }
        }
      } else {
        bf16* op = out + (size_t)rbase * NW + ncol;
        if (fullstrip) {
#pragma unroll
          for (int i = 0; i < 8; ++i) op[i * NW] = (bf16)vv[i];  // imm-offset stores
        } else {
#pragma unroll
          for (int i = 0; i < 8; ++i)
            if (rbase + i < M) op[i * NW] = (bf16)vv[i];
        }
      }
    }
  }
}

// ---------------------------------------------------------------------------
// Final sigmoid heads
// ---------------------------------------------------------------------------
__global__ void k_head_x(const bf16* __restrict__ xbf, const float* __restrict__ W,
                         const float* __restrict__ b, float* __restrict__ out, int N) {
  int n = blockIdx.x * blockDim.x + threadIdx.x;
  if (n >= N) return;
  const bf16* p = xbf + (size_t)n * 16;
  float s = b[0];
#pragma unroll
  for (int i = 0; i < 16; ++i) s += (float)p[i] * W[i];
  out[n] = 1.0f / (1.0f + expf(-s));
}

__global__ void k_head_e(const bf16* __restrict__ eabf, const float* __restrict__ W,
                         const float* __restrict__ b, float* __restrict__ out, int E) {
  int e = blockIdx.x * blockDim.x + threadIdx.x;
  if (e >= E) return;
  const bf16* p = eabf + (size_t)e * 32;
  float s = b[0];
#pragma unroll
  for (int i = 0; i < 19; ++i) s += (float)p[i] * W[i];
  out[e] = 1.0f / (1.0f + expf(-s));
}

// ---------------------------------------------------------------------------
// Host orchestration
// ---------------------------------------------------------------------------
static void run_block(const bf16* A, int M, int K, int KPAD, int NOUT, int NTOT,
                      const float* g, const float* b, const float* W, const float* c,
                      float* stats, bf16* Wb, float* cb, hipStream_t s) {
  hipMemsetAsync(stats, 0, 2 * SSTR * sizeof(float), s);
  k_colstats<<<1024, 256, 0, s>>>(A, (long long)M, KPAD, stats);
  k_prep<<<(NTOT + 63) / 64, 64, 0, s>>>(stats, 1.0f / (float)M, g, b, W, c,
                                         K, KPAD, NOUT, NTOT, Wb, cb);
}

template <int KT, int NT, int NOUT, bool RELU, bool SCATTER>
static void launch_gemm(const bf16* A, int M, const bf16* Wb, const float* cb,
                        bf16* out, const int* sidx, float* agg, hipStream_t s) {
  // B-fragment reuse factor, bounded by register pressure (A frags = MT*KT*8 VGPRs)
  constexpr int MT = (KT <= 2) ? 4 : (KT <= 5) ? 2 : 1;
  int tiles = (M + MT * 16 - 1) / (MT * 16);
  int blocks = (tiles + 7) / 8;  // 8 waves (256 threads) per block
  k_gemm<MT, KT, NT, NOUT, RELU, SCATTER><<<blocks, 256, 0, s>>>(A, M, Wb, cb, out, sidx, agg);
}

template <int L>
static void run_round(void* const* d_in, int pb, int N, int E,
                      bf16* xbf, bf16* eabf, bf16* eA, bf16* eB, bf16* nA, bf16* nB,
                      float* agg, float* stats, bf16* Wb, float* cb,
                      const int* row, const int* col, hipStream_t s) {
  auto P = [&](int i) { return (const float*)d_in[i]; };
  constexpr int S    = 16 + L;
  constexpr int SPAD = ((S + 31) / 32) * 32;
  constexpr int KTL  = L / 32, NTL = L / 16;
  constexpr int KTS  = SPAD / 32, NTS = SPAD / 16;

  // ---- Edge MLP: concat(51) -> L -> L -> 19 ----
  k_concat_edge<<<(E + 255) / 256, 256, 0, s>>>(xbf, eabf, row, col, eA, E);
  run_block(eA, E, 51, 64, L, L, P(pb + 0), P(pb + 1), P(pb + 2), P(pb + 3), stats, Wb, cb, s);
  launch_gemm<2, NTL, L, true, false>(eA, E, Wb, cb, eB, nullptr, nullptr, s);
  run_block(eB, E, L, L, L, L, P(pb + 4), P(pb + 5), P(pb + 6), P(pb + 7), stats, Wb, cb, s);
  launch_gemm<KTL, NTL, L, true, false>(eB, E, Wb, cb, eA, nullptr, nullptr, s);
  run_block(eA, E, L, L, 19, 32, P(pb + 8), P(pb + 9), P(pb + 10), P(pb + 11), stats, Wb, cb, s);
  launch_gemm<KTL, 2, 19, false, false>(eA, E, Wb, cb, eabf, nullptr, nullptr, s);

  // ---- Node1 MLP: concat(35) -> L -> L -> L, scatter-add over col ----
  k_concat_node1<<<(E + 255) / 256, 256, 0, s>>>(xbf, eabf, row, eB, E);
  run_block(eB, E, 35, 64, L, L, P(pb + 12), P(pb + 13), P(pb + 14), P(pb + 15), stats, Wb, cb, s);
  launch_gemm<2, NTL, L, true, false>(eB, E, Wb, cb, eA, nullptr, nullptr, s);
  run_block(eA, E, L, L, L, L, P(pb + 16), P(pb + 17), P(pb + 18), P(pb + 19), stats, Wb, cb, s);
  launch_gemm<KTL, NTL, L, true, false>(eA, E, Wb, cb, eB, nullptr, nullptr, s);
  run_block(eB, E, L, L, L, L, P(pb + 20), P(pb + 21), P(pb + 22), P(pb + 23), stats, Wb, cb, s);
  hipMemsetAsync(agg, 0, (size_t)N * L * sizeof(float), s);
  launch_gemm<KTL, NTL, L, false, true>(eB, E, Wb, cb, nullptr, col, agg, s);

  // ---- Node2 MLP: concat(S) -> S -> S -> 16 ----
  k_concat_node2<<<(N + 255) / 256, 256, 0, s>>>(xbf, agg, nA, N, L, SPAD);
  run_block(nA, N, S, SPAD, S, SPAD, P(pb + 24), P(pb + 25), P(pb + 26), P(pb + 27), stats, Wb, cb, s);
  launch_gemm<KTS, NTS, S, true, false>(nA, N, Wb, cb, nB, nullptr, nullptr, s);
  run_block(nB, N, S, SPAD, S, SPAD, P(pb + 28), P(pb + 29), P(pb + 30), P(pb + 31), stats, Wb, cb, s);
  launch_gemm<KTS, NTS, S, true, false>(nB, N, Wb, cb, nA, nullptr, nullptr, s);
  run_block(nA, N, S, SPAD, 16, 16, P(pb + 32), P(pb + 33), P(pb + 34), P(pb + 35), stats, Wb, cb, s);
  launch_gemm<KTS, 1, 16, false, false>(nA, N, Wb, cb, xbf, nullptr, nullptr, s);
}

extern "C" void kernel_launch(void* const* d_in, const int* in_sizes, int n_in,
                              void* d_out, int out_size, void* d_ws, size_t ws_size,
                              hipStream_t stream) {
  (void)n_in; (void)out_size; (void)ws_size;
  const int N = in_sizes[0] / 16;   // 50000
  const int E = in_sizes[1] / 2;    // 400000
  const float* x0  = (const float*)d_in[0];
  const int*   ei  = (const int*)d_in[1];
  const float* ea0 = (const float*)d_in[2];
  const int* row = ei;
  const int* col = ei + E;

  // Param flattening assumption: insertion order of the params dict
  // (ml1, ml2, ml3, x_lin, e_lin), each ml = {edge, node1, node2} x 3 blocks
  // of (g, b, W, c).  Base index 5 after x, edge_index, edge_attr, u, batch.
  const int PB_ML1 = 5, PB_ML2 = 5 + 36, PB_ML3 = 5 + 72;
  const float* Wx = (const float*)d_in[113];
  const float* bx = (const float*)d_in[114];
  const float* We = (const float*)d_in[115];
  const float* be = (const float*)d_in[116];

  // Workspace layout
  unsigned char* w = (unsigned char*)d_ws;
  size_t off = 0;
  auto alloc = [&](size_t bytes) {
    void* p = w + off;
    off = (off + bytes + 255) & ~(size_t)255;
    return p;
  };
  bf16* xbf   = (bf16*)alloc((size_t)N * 16 * sizeof(bf16));
  bf16* eabf  = (bf16*)alloc((size_t)E * 32 * sizeof(bf16));
  bf16* eA    = (bf16*)alloc((size_t)E * 288 * sizeof(bf16));
  bf16* eB    = (bf16*)alloc((size_t)E * 288 * sizeof(bf16));
  bf16* nA    = (bf16*)alloc((size_t)N * 288 * sizeof(bf16));
  bf16* nB    = (bf16*)alloc((size_t)N * 288 * sizeof(bf16));
  float* agg  = (float*)alloc((size_t)N * 256 * sizeof(float));
  float* stats = (float*)alloc(2 * SSTR * sizeof(float));
  bf16* Wb    = (bf16*)alloc((size_t)288 * 288 * sizeof(bf16));
  float* cb   = (float*)alloc(288 * sizeof(float));

  // Initial bf16 conversion (padded)
  {
    long long t = (long long)N * 16;
    k_cvt_pad<<<(unsigned)((t + 255) / 256), 256, 0, stream>>>(x0, xbf, N, 16, 16);
  }
  {
    long long t = (long long)E * 32;
    k_cvt_pad<<<(unsigned)((t + 255) / 256), 256, 0, stream>>>(ea0, eabf, E, 19, 32);
  }

  run_round<64 >(d_in, PB_ML1, N, E, xbf, eabf, eA, eB, nA, nB, agg, stats, Wb, cb, row, col, stream);
  run_round<256>(d_in, PB_ML2, N, E, xbf, eabf, eA, eB, nA, nB, agg, stats, Wb, cb, row, col, stream);
  run_round<128>(d_in, PB_ML3, N, E, xbf, eabf, eA, eB, nA, nB, agg, stats, Wb, cb, row, col, stream);

  float* out = (float*)d_out;
  k_head_x<<<(N + 255) / 256, 256, 0, stream>>>(xbf, Wx, bx, out, N);
  k_head_e<<<(E + 255) / 256, 256, 0, stream>>>(eabf, We, be, out + N, E);
}
